// BiHgru2d_65446711656542
// MI455X (gfx1250) — compile-verified, matrix operand order checked by reading, not run on hardware
//
#include <hip/hip_runtime.h>

// ---------------------------------------------------------------------------
// BiHgru2d for MI455X (gfx1250): fused bf16-WMMA GEMMs + LDS-fused scans
// H=128, W=128, B=8, D=128. M = H*W*B = 131072 flattened rows.
//
// Roofline: ~30 GFLOP total, dominated by ~1 GB HBM traffic @ 23.3 TB/s.
// GEMMs run on v_wmma_f32_16x16x32_bf16 (fp32 accum); scans are fused
// fwd+rev per axis with 160 KB LDS staging (320 KB/WGP available on CDNA5)
// to halve their global traffic.
// ---------------------------------------------------------------------------

#define HH 128
#define WW 128
#define BB 8
#define DD 128
#define MM (HH * WW * BB)          // 131072
#define NCAT 640                   // 256 (in) + 256 (gate) + 128 (lambda)

typedef unsigned short u16;
typedef __attribute__((ext_vector_type(16))) __bf16 v16bf;
typedef __attribute__((ext_vector_type(8)))  float  v8f;

union FragBF {
    uint4  u[2];   // 2 x 16B = 32B
    v16bf  v;      // 16 x bf16 = 32B
};

__device__ __forceinline__ u16 f2bf(float f) {
    union { float f; unsigned u; } c; c.f = f;
    unsigned u = c.u;
    u += 0x7fffu + ((u >> 16) & 1u);       // round-to-nearest-even
    return (u16)(u >> 16);
}

__device__ __forceinline__ float siluf(float v)    { return v / (1.f + __expf(-v)); }
__device__ __forceinline__ float sigmoidf_(float v){ return 1.f / (1.f + __expf(-v)); }

// ---------------------------------------------------------------------------
// Prep: pack weights into WMMA-friendly bf16 layouts.
//  wcat : [NCAT][128]  column-major-per-N (each output column's K contiguous)
//  woutT: [128][256]   column-major-per-N for the output GEMM (K = 256)
//  bcat : [NCAT] fp32 concatenated biases
// ---------------------------------------------------------------------------
__global__ void prep_weights(const float* __restrict__ W_in,
                             const float* __restrict__ W_gate,
                             const float* __restrict__ W_lambda,
                             const float* __restrict__ b_in,
                             const float* __restrict__ b_gate,
                             const float* __restrict__ b_lambda,
                             const float* __restrict__ W_out,
                             u16* __restrict__ wcat,
                             u16* __restrict__ woutT,
                             float* __restrict__ bcat) {
    int i = blockIdx.x * blockDim.x + threadIdx.x;
    if (i < NCAT * DD) {                       // wcat
        int n = i / DD, k = i % DD;
        float v = (n < 256) ? W_in[k * 256 + n]
                : (n < 512) ? W_gate[k * 256 + (n - 256)]
                            : W_lambda[k * DD + (n - 512)];
        wcat[i] = f2bf(v);
    }
    int i2 = i - NCAT * DD;
    if (i2 >= 0 && i2 < DD * 256) {            // woutT: n = i2/256, k = i2%256
        int n = i2 / 256, k = i2 % 256;
        woutT[i2] = f2bf(W_out[k * DD + n]);
    }
    int i3 = i - NCAT * DD - DD * 256;
    if (i3 >= 0 && i3 < NCAT) {
        bcat[i3] = (i3 < 256) ? b_in[i3]
                 : (i3 < 512) ? b_gate[i3 - 256]
                              : b_lambda[i3 - 512];
    }
}

__global__ void cvt_x_bf16(const float* __restrict__ x, u16* __restrict__ xbf, int n) {
    int i = blockIdx.x * blockDim.x + threadIdx.x;
    if (i < n) xbf[i] = f2bf(x[i]);
}

// ---------------------------------------------------------------------------
// Fused wide GEMM: [M x 128] @ [128 x 640] -> inp (silu), gate (silu),
// lam (sigmoid).  One wave per 16-row tile x 64-col group; bf16 WMMA,
// fp32 accumulation.  A/B fragments: two global_load_b128 per lane, K-order
// permuted identically in A and B (dot product invariant).  xbf (33.5 MB)
// and wcat (160 KB) both stay resident in the 192 MB L2 across the 10
// column-group passes, so HBM sees them ~once.
// ---------------------------------------------------------------------------
__global__ void gemm_act(const u16* __restrict__ xbf,    // [M][128] row-major
                         const u16* __restrict__ wcat,   // [640][128] N-major
                         const float* __restrict__ bcat, // [640]
                         float* __restrict__ inp,        // [M][256]
                         float* __restrict__ gate,       // [M][256]
                         float* __restrict__ lam)        // [M][128]
{
    const int lane  = threadIdx.x & 31;
    const int wave  = threadIdx.x >> 5;
    const int mtile = blockIdx.x * 8 + wave;        // 8 waves per block
    const int m0    = mtile * 16;
    const int n0    = blockIdx.y * 64;
    const int rsel  = lane & 15;
    const int koff  = (lane >> 4) * 8;              // K half-select per lane group

    const u16* arow = xbf + (size_t)(m0 + rsel) * DD;

    v8f acc0 = {}, acc1 = {}, acc2 = {}, acc3 = {};
    v8f* accs[4] = {&acc0, &acc1, &acc2, &acc3};

    for (int k0 = 0; k0 < DD; k0 += 32) {
        FragBF a;
        a.u[0] = *(const uint4*)(arow + k0 + koff);
        a.u[1] = *(const uint4*)(arow + k0 + koff + 16);
#pragma unroll
        for (int t = 0; t < 4; ++t) {
            const u16* bcol = wcat + (size_t)(n0 + t * 16 + rsel) * DD;
            FragBF b;
            b.u[0] = *(const uint4*)(bcol + k0 + koff);
            b.u[1] = *(const uint4*)(bcol + k0 + koff + 16);
            *accs[t] = __builtin_amdgcn_wmma_f32_16x16x32_bf16(
                false, a.v, false, b.v, (short)0, *accs[t], false, false);
        }
    }

#pragma unroll
    for (int t = 0; t < 4; ++t) {
        const int ncol = n0 + t * 16 + rsel;
        const float bias = bcat[ncol];
        const v8f c = *accs[t];
#pragma unroll
        for (int r = 0; r < 8; ++r) {
            const int row = m0 + r + 8 * (lane >> 4);
            const float v = c[r] + bias;
            if (ncol < 256)      inp [(size_t)row * 256 + ncol]          = siluf(v);
            else if (ncol < 512) gate[(size_t)row * 256 + (ncol - 256)]  = siluf(v);
            else                 lam [(size_t)row * DD  + (ncol - 512)]  = sigmoidf_(v);
        }
    }
}

// ---------------------------------------------------------------------------
// Fused fwd+rev complex gated scan along one axis.
//
// Block = 64 threads; each block owns 64 adjacent channels (d) of one
// (o, b) line, sequence length 128.  ir/ii/lam are staged in LDS once
// (3 x 32 KB), the reverse scan (theta_r) writes its trajectory into a
// 2 x 32 KB LDS result buffer, then the forward scan (theta_f) combines
// both and emits ONE global write per element (init) or one RMW (add).
// 160 KB LDS per workgroup -- legal on CDNA5 (320 KB per WGP), requested
// dynamically.  Each thread only touches its own LDS column -> no barriers.
// global_prefetch_b8 covers the strided staging stream 8 steps ahead.
// ---------------------------------------------------------------------------
__global__ void scan_axis_fused(const float* __restrict__ inp,
                                const float* __restrict__ lam,
                                const float* __restrict__ theta_f,
                                const float* __restrict__ theta_r,
                                float* __restrict__ hr_acc,
                                float* __restrict__ hi_acc,
                                int axis, int init)
{
    extern __shared__ float smem[];
    float* s_ir = smem;                    // [128][64]
    float* s_ii = smem + 128 * 64;         // [128][64]
    float* s_lm = smem + 2 * 128 * 64;     // [128][64]
    float* s_hr = smem + 3 * 128 * 64;     // [128][64] reverse-scan result
    float* s_hi = smem + 4 * 128 * 64;     // [128][64]

    const int t  = threadIdx.x;            // 0..63: channel within group
    const int gb = blockIdx.x;             // 0..2047
    const int dh = gb & 1;                 // which half of D
    const int b  = (gb >> 1) & (BB - 1);
    const int o  = gb >> 4;                // 0..127: the fixed coordinate
    const int d  = dh * 64 + t;

    // ---- stage ir / ii / lam for all 128 positions ----
#pragma unroll 4
    for (int p = 0; p < 128; ++p) {
        const int h_ = (axis == 0) ? p : o;
        const int w_ = (axis == 0) ? o : p;
        const int idx = (h_ * WW + w_) * BB + b;
        if (p + 8 < 128) {
            const int h2 = (axis == 0) ? (p + 8) : o;
            const int w2 = (axis == 0) ? o : (p + 8);
            const int idx2 = (h2 * WW + w2) * BB + b;
            __builtin_prefetch(lam + (size_t)idx2 * DD + d, 0, 0);
            __builtin_prefetch(inp + (size_t)idx2 * 256 + 2 * d, 0, 0);
        }
        const float l  = lam[(size_t)idx * DD + d];
        const float ie = inp[(size_t)idx * 256 + 2 * d];
        const float io = inp[(size_t)idx * 256 + 2 * d + 1];
        const float omL = 1.f - l;
        s_lm[p * 64 + t] = l;
        s_ir[p * 64 + t] = omL * ie;
        s_ii[p * 64 + t] = omL * io;
    }

    // ---- reverse scan (theta_r), trajectory into LDS ----
    {
        const float th = theta_r[d];
        const float ct = __cosf(th), st = __sinf(th);
        float hr = 0.f, hi = 0.f;
#pragma unroll 4
        for (int s = 0; s < 128; ++s) {
            const int p = 127 - s;
            const float l  = s_lm[p * 64 + t];
            const float gr = l * ct, gi = l * st;
            const float nr = gr * hr - gi * hi + s_ir[p * 64 + t];
            const float ni = gr * hi + gi * hr + s_ii[p * 64 + t];
            hr = nr; hi = ni;
            s_hr[p * 64 + t] = hr;
            s_hi[p * 64 + t] = hi;
        }
    }

    // ---- forward scan (theta_f), combine + single global write ----
    {
        const float th = theta_f[d];
        const float ct = __cosf(th), st = __sinf(th);
        float hr = 0.f, hi = 0.f;
#pragma unroll 4
        for (int p = 0; p < 128; ++p) {
            const float l  = s_lm[p * 64 + t];
            const float gr = l * ct, gi = l * st;
            const float nr = gr * hr - gi * hi + s_ir[p * 64 + t];
            const float ni = gr * hi + gi * hr + s_ii[p * 64 + t];
            hr = nr; hi = ni;
            const float orv = hr + s_hr[p * 64 + t];
            const float oiv = hi + s_hi[p * 64 + t];
            const int h_ = (axis == 0) ? p : o;
            const int w_ = (axis == 0) ? o : p;
            const size_t a = (size_t)((h_ * WW + w_) * BB + b) * DD + d;
            if (init) { hr_acc[a] = orv;  hi_acc[a] = oiv;  }
            else      { hr_acc[a] += orv; hi_acc[a] += oiv; }
        }
    }
}

// ---------------------------------------------------------------------------
// Fused LayerNorm(256) + gate + output GEMM [16x256]@[256x128] via WMMA.
// Feature tile staged in LDS as bf16 in the A-fragment layout; 8 waves each
// own one 16-column N tile.
// ---------------------------------------------------------------------------
__global__ void final_kernel(const float* __restrict__ hr_acc,
                             const float* __restrict__ hi_acc,
                             const float* __restrict__ gate,
                             const u16*  __restrict__ woutT,  // [128][256] N-major
                             const float* __restrict__ ln_w,
                             const float* __restrict__ ln_b,
                             const float* __restrict__ b_out,
                             float* __restrict__ out)         // [M][128]
{
    __shared__ float fs[16 * 256];
    __shared__ __align__(16) u16 fa[16 * 256];
    __shared__ float mu_s[16], rs_s[16];

    const int tid = threadIdx.x;
    const int m0  = blockIdx.x * 16;

    // 1. build raw feature tile: [16 rows][256 = hr|hi]
    for (int e = tid; e < 16 * 256; e += 256) {
        const int r = e >> 8, c = e & 255;
        const int row = m0 + r;
        fs[e] = (c < DD) ? hr_acc[(size_t)row * DD + c]
                         : hi_acc[(size_t)row * DD + (c - DD)];
    }
    __syncthreads();

    // 2. per-row mean / rstd
    if (tid < 16) {
        float s = 0.f, s2 = 0.f;
        for (int c = 0; c < 256; ++c) { const float f = fs[tid * 256 + c]; s += f; s2 += f * f; }
        const float mu  = s * (1.f / 256.f);
        const float var = s2 * (1.f / 256.f) - mu * mu;
        mu_s[tid] = mu;
        rs_s[tid] = rsqrtf(var + 1e-5f);
    }
    __syncthreads();

    // 3. normalize, scale/shift, gate, convert to bf16 A tile
    for (int e = tid; e < 16 * 256; e += 256) {
        const int r = e >> 8, c = e & 255;
        const int row = m0 + r;
        float f = (fs[e] - mu_s[r]) * rs_s[r] * ln_w[c] + ln_b[c];
        f *= gate[(size_t)row * 256 + c];
        fa[e] = f2bf(f);
    }
    __syncthreads();

    // 4. WMMA output GEMM: K = 256 in 8 steps of 32
    const int lane = tid & 31, wave = tid >> 5;
    const int rsel = lane & 15, koff = (lane >> 4) * 8;
    const int n0   = wave * 16;

    v8f acc = {};
    for (int k0 = 0; k0 < 256; k0 += 32) {
        FragBF a;
        const u16* ap = fa + rsel * 256 + k0 + koff;
        a.u[0] = *(const uint4*)ap;
        a.u[1] = *(const uint4*)(ap + 16);
        FragBF b;
        const u16* bp = woutT + (size_t)(n0 + rsel) * 256 + k0 + koff;
        b.u[0] = *(const uint4*)bp;
        b.u[1] = *(const uint4*)(bp + 16);
        acc = __builtin_amdgcn_wmma_f32_16x16x32_bf16(
            false, a.v, false, b.v, (short)0, acc, false, false);
    }

    const float bias = b_out[n0 + rsel];
#pragma unroll
    for (int r = 0; r < 8; ++r) {
        const int row = m0 + r + 8 * (lane >> 4);
        out[(size_t)row * DD + n0 + rsel] = acc[r] + bias;
    }
}

// ---------------------------------------------------------------------------
// Host side
// ---------------------------------------------------------------------------
extern "C" void kernel_launch(void* const* d_in, const int* in_sizes, int n_in,
                              void* d_out, int out_size, void* d_ws, size_t ws_size,
                              hipStream_t stream) {
    const float* x        = (const float*)d_in[0];
    const float* W_in     = (const float*)d_in[1];
    const float* b_in     = (const float*)d_in[2];
    const float* W_lambda = (const float*)d_in[3];
    const float* b_lambda = (const float*)d_in[4];
    const float* theta_f  = (const float*)d_in[5];
    const float* theta_r  = (const float*)d_in[6];
    const float* W_gate   = (const float*)d_in[7];
    const float* b_gate   = (const float*)d_in[8];
    const float* W_out    = (const float*)d_in[9];
    const float* b_out    = (const float*)d_in[10];
    const float* ln_w     = (const float*)d_in[11];
    const float* ln_b     = (const float*)d_in[12];
    float* out = (float*)d_out;

    char* ws = (char*)d_ws;
    size_t off = 0;
    auto carve = [&](size_t bytes) -> void* {
        void* p = ws + off;
        off = (off + bytes + 255) & ~(size_t)255;
        return p;
    };

    u16*   xbf   = (u16*)  carve((size_t)MM * DD * 2);       // 33.6 MB
    u16*   wcat  = (u16*)  carve((size_t)NCAT * DD * 2);     // 160 KB
    u16*   woutT = (u16*)  carve((size_t)DD * 256 * 2);      // 64 KB
    float* bcat  = (float*)carve((size_t)NCAT * 4);
    float* inp   = (float*)carve((size_t)MM * 256 * 4);      // 134 MB
    float* gate  = (float*)carve((size_t)MM * 256 * 4);      // 134 MB
    float* lam   = (float*)carve((size_t)MM * DD * 4);       // 67 MB
    float* hr    = (float*)carve((size_t)MM * DD * 4);       // 67 MB
    float* hi    = (float*)carve((size_t)MM * DD * 4);       // 67 MB
    (void)ws_size; (void)n_in; (void)in_sizes; (void)out_size;

    // 1. weight packing + x conversion
    {
        int total = NCAT * DD + DD * 256 + NCAT;
        prep_weights<<<(total + 255) / 256, 256, 0, stream>>>(
            W_in, W_gate, W_lambda, b_in, b_gate, b_lambda, W_out,
            wcat, woutT, bcat);
        int nx = MM * DD;
        cvt_x_bf16<<<(nx + 255) / 256, 256, 0, stream>>>(x, xbf, nx);
    }

    // 2. fused wide GEMM -> inp / gate / lam
    {
        dim3 grid(MM / 16 / 8, NCAT / 64);   // (1024, 10)
        gemm_act<<<grid, 256, 0, stream>>>(xbf, wcat, bcat, inp, gate, lam);
    }

    // 3. two fused (fwd+rev) scans, one per axis; 160 KB dynamic LDS each
    {
        const int blocks = (WW * BB * DD) / 64;  // 2048
        const size_t lds = 5 * 128 * 64 * sizeof(float);  // 160 KB
        scan_axis_fused<<<blocks, 64, lds, stream>>>(
            inp, lam, theta_f, theta_r, hr, hi, /*axis=*/0, /*init=*/1);
        scan_axis_fused<<<blocks, 64, lds, stream>>>(
            inp, lam, theta_f, theta_r, hr, hi, /*axis=*/1, /*init=*/0);
    }

    // 4. LayerNorm + gate + output GEMM
    final_kernel<<<MM / 16, 256, 0, stream>>>(hr, hi, gate, woutT,
                                              ln_w, ln_b, b_out, out);
}
